// GAT_28716151341635
// MI455X (gfx1250) — matrix operand order repaired and checked by reading, not compiled, not run on hardware
//
#include <hip/hip_runtime.h>
#include <stdint.h>

#define NNODES 50000
#define NEDGES 800000
#define ETOT   (NEDGES + NNODES)   // edges + self loops
#define FIN    512
#define H1DIM  256
#define COUT   64
#define NEG_SLOPE 0.2f
#define EPSV   1e-16f

typedef __attribute__((ext_vector_type(16))) __bf16 v16bf;
typedef __attribute__((ext_vector_type(8)))  __bf16 v8bf;
typedef __attribute__((ext_vector_type(8)))  float  v8f;

// ---------- helpers ----------
static __device__ __forceinline__ unsigned f2bf_bits(float f) {
  unsigned u = __float_as_uint(f);
  unsigned r = ((u >> 16) & 1u) + 0x7FFFu;     // round-to-nearest-even
  return (u + r) >> 16;
}
static __device__ __forceinline__ unsigned f2ord(float f) {  // monotonic f32 -> u32
  unsigned u = __float_as_uint(f);
  return (u & 0x80000000u) ? ~u : (u | 0x80000000u);
}
static __device__ __forceinline__ float ord2f(unsigned o) {
  unsigned u = (o & 0x80000000u) ? (o & 0x7FFFFFFFu) : ~o;
  return __uint_as_float(u);
}
// edge_index arrives as int32 (JAX default x64-disabled downgrades jnp.int64)
static __device__ __forceinline__ void edge_nodes(const int* __restrict__ ei,
                                                  int e, int& s, int& d) {
  if (e < NEDGES) { s = ei[e]; d = ei[NEDGES + e]; }
  else            { s = e - NEDGES; d = e - NEDGES; }   // self loop
}

// ---------- vectorized f32 -> bf16 (8 elems/thread, b128 in/out) ----------
__global__ void k_cvt_bf16x8(const float* __restrict__ in,
                             unsigned short* __restrict__ out, int n8) {
  int i = blockIdx.x * blockDim.x + threadIdx.x;   // index of 8-element group
  if (i >= n8) return;
  const float4* p = (const float4*)(in + (size_t)i * 8);
  float4 a = p[0], b = p[1];
  uint4 o;
  o.x = f2bf_bits(a.x) | (f2bf_bits(a.y) << 16);
  o.y = f2bf_bits(a.z) | (f2bf_bits(a.w) << 16);
  o.z = f2bf_bits(b.x) | (f2bf_bits(b.y) << 16);
  o.w = f2bf_bits(b.z) | (f2bf_bits(b.w) << 16);
  *(uint4*)(out + (size_t)i * 8) = o;
}

// ---------- Wt[n][k] = W[k][n], f32 -> bf16 (W is [K][H]) ----------
__global__ void k_transpose_bf16(const float* __restrict__ W,
                                 unsigned short* __restrict__ Wt, int K, int H) {
  int i = blockIdx.x * blockDim.x + threadIdx.x;
  if (i >= K * H) return;
  int k = i % K, n = i / K;                 // write Wt contiguously
  Wt[(size_t)n * K + k] = (unsigned short)f2bf_bits(W[(size_t)k * H + n]);
}

// ---------- WMMA GEMM: C[M][Nout] = A[M][K](bf16,row-major) x Bt[Nout][K]^T ----------
// one wave per block; wave computes one 16-row stripe x 64 columns (4 WMMA N-tiles)
__global__ void __launch_bounds__(32)
k_wmma_gemm(const __bf16* __restrict__ A, const __bf16* __restrict__ Bt,
            float* __restrict__ C, int M, int K, int Nout) {
  const int lane  = threadIdx.x & 31;
  const int mbase = blockIdx.x * 16;
  const int nbase = blockIdx.y * 64;
  const int row   = mbase + (lane & 15);
  const int kb    = (lane >> 4) * 8;     // unused; A fragment also uses kh
  const int kh    = (lane >> 4) * 16;    // B-fragment K half offset
  const int col   = nbase + (lane & 15);

  v8f acc0 = {}, acc1 = {}, acc2 = {}, acc3 = {};
  for (int kk = 0; kk < K; kk += 32) {
    const __bf16* arow = A + (size_t)row * K + kk;
    v16bf a = *(const v16bf*)(arow + kh);

    v16bf b0 = *(const v16bf*)(Bt + (size_t)(col +  0) * K + kk + kh);
    v16bf b1 = *(const v16bf*)(Bt + (size_t)(col + 16) * K + kk + kh);
    v16bf b2 = *(const v16bf*)(Bt + (size_t)(col + 32) * K + kk + kh);
    v16bf b3 = *(const v16bf*)(Bt + (size_t)(col + 48) * K + kk + kh);

    acc0 = __builtin_amdgcn_wmma_f32_16x16x32_bf16(false, a, false, b0, (short)0, acc0, false, false);
    acc1 = __builtin_amdgcn_wmma_f32_16x16x32_bf16(false, a, false, b1, (short)0, acc1, false, false);
    acc2 = __builtin_amdgcn_wmma_f32_16x16x32_bf16(false, a, false, b2, (short)0, acc2, false, false);
    acc3 = __builtin_amdgcn_wmma_f32_16x16x32_bf16(false, a, false, b3, (short)0, acc3, false, false);
  }

  const int crow0 = mbase + (lane >> 4) * 8;
  const int ccol  = nbase + (lane & 15);
#pragma unroll
  for (int r = 0; r < 8; ++r) {
    float* crow = C + (size_t)(crow0 + r) * Nout + ccol;
    crow[ 0] = acc0[r];
    crow[16] = acc1[r];
    crow[32] = acc2[r];
    crow[48] = acc3[r];
  }
}

// ---------- per-node attention dot products (one wave per node) ----------
__global__ void k_alphas(const float* __restrict__ h, const float* __restrict__ av_s,
                         const float* __restrict__ av_d, float* __restrict__ osrc,
                         float* __restrict__ odst, int N, int F) {
  int wave = (blockIdx.x * blockDim.x + threadIdx.x) >> 5;
  int lane = threadIdx.x & 31;
  if (wave >= N) return;
  const float* row = h + (size_t)wave * F;
  float ss = 0.f, sd = 0.f;
  for (int f = lane; f < F; f += 32) {
    float v = row[f];
    ss += v * av_s[f];
    sd += v * av_d[f];
  }
#pragma unroll
  for (int off = 16; off > 0; off >>= 1) {
    ss += __shfl_xor(ss, off, 32);
    sd += __shfl_xor(sd, off, 32);
  }
  if (lane == 0) { osrc[wave] = ss; odst[wave] = sd; }
}

// ---------- edge pass 1: segment max (ordered-uint atomicMax) ----------
__global__ void k_edge_max(const int* __restrict__ ei, const float* __restrict__ as,
                           const float* __restrict__ ad, unsigned* __restrict__ smax) {
  int e = blockIdx.x * blockDim.x + threadIdx.x;
  if (e >= ETOT) return;
  int s, d; edge_nodes(ei, e, s, d);
  float v = as[s] + ad[d];
  v = v > 0.f ? v : NEG_SLOPE * v;
  atomicMax(&smax[d], f2ord(v));
}

// ---------- edge pass 2: exp + segment sum ----------
__global__ void k_edge_exp(const int* __restrict__ ei, const float* __restrict__ as,
                           const float* __restrict__ ad, const unsigned* __restrict__ smax,
                           float* __restrict__ ssum, float* __restrict__ exv) {
  int e = blockIdx.x * blockDim.x + threadIdx.x;
  if (e >= ETOT) return;
  int s, d; edge_nodes(ei, e, s, d);
  float v = as[s] + ad[d];
  v = v > 0.f ? v : NEG_SLOPE * v;
  float ex = __expf(v - ord2f(smax[d]));
  exv[e] = ex;
  atomicAdd(&ssum[d], ex);
}

// ---------- edge pass 3: weighted scatter-add (one wave per edge) ----------
__global__ void k_edge_agg(const int* __restrict__ ei, const float* __restrict__ h,
                           const float* __restrict__ exv, const float* __restrict__ ssum,
                           float* __restrict__ out, int F) {
  int wave = (blockIdx.x * blockDim.x + threadIdx.x) >> 5;
  int lane = threadIdx.x & 31;
  if (wave >= ETOT) return;
  int s, d; edge_nodes(ei, wave, s, d);
  float w = exv[wave] / (ssum[d] + EPSV);
  const float* hs = h   + (size_t)s * F;
  float*       od = out + (size_t)d * F;
  for (int f = lane; f < F; f += 32)
    atomicAdd(&od[f], hs[f] * w);
}

// ---------- bias (+ optional ReLU) ----------
__global__ void k_bias_act(float* __restrict__ y, const float* __restrict__ b,
                           int total, int F, int do_relu) {
  int i = blockIdx.x * blockDim.x + threadIdx.x;
  if (i >= total) return;
  float v = y[i] + b[i % F];
  if (do_relu) v = fmaxf(v, 0.f);
  y[i] = v;
}

extern "C" void kernel_launch(void* const* d_in, const int* in_sizes, int n_in,
                              void* d_out, int out_size, void* d_ws, size_t ws_size,
                              hipStream_t stream) {
  const float* x   = (const float*)d_in[0];
  const int*   ei  = (const int*)d_in[1];     // int32 edge_index (2, E)
  const float* W1  = (const float*)d_in[2];
  const float* as1 = (const float*)d_in[3];
  const float* ad1 = (const float*)d_in[4];
  const float* b1  = (const float*)d_in[5];
  const float* W2  = (const float*)d_in[6];
  const float* as2 = (const float*)d_in[7];
  const float* ad2 = (const float*)d_in[8];
  const float* b2  = (const float*)d_in[9];
  float* out = (float*)d_out;

  char* ws = (char*)d_ws;
  size_t off = 0;
  auto take = [&](size_t bytes) -> char* {
    char* p = ws + off;
    off = (off + bytes + 255) & ~(size_t)255;
    return p;
  };
  unsigned short* xh  = (unsigned short*)take((size_t)NNODES * FIN * 2);   // bf16 acts
  unsigned short* Wt  = (unsigned short*)take((size_t)FIN * H1DIM * 2);    // bf16 W^T
  float*    hd   = (float*)take((size_t)NNODES * H1DIM * 4);               // dense GEMM out
  float*    agg  = (float*)take((size_t)NNODES * H1DIM * 4);               // layer-1 agg out
  float*    asn  = (float*)take((size_t)NNODES * 4);
  float*    adn  = (float*)take((size_t)NNODES * 4);
  unsigned* smax = (unsigned*)take((size_t)NNODES * 4);
  float*    ssum = (float*)take((size_t)NNODES * 4);
  float*    exv  = (float*)take((size_t)ETOT * 4);

  const int eb  = (ETOT + 255) / 256;          // thread-per-edge blocks
  const int ewb = (ETOT + 7) / 8;              // wave-per-edge blocks (256 thr = 8 waves)
  const int nwb = (NNODES + 7) / 8;            // wave-per-node blocks

  // ============ Layer 1 ============
  { int n8 = (NNODES * FIN) / 8;
    k_cvt_bf16x8<<<(n8 + 255) / 256, 256, 0, stream>>>(x, xh, n8); }
  { int n = FIN * H1DIM;
    k_transpose_bf16<<<(n + 255) / 256, 256, 0, stream>>>(W1, Wt, FIN, H1DIM); }
  k_wmma_gemm<<<dim3(NNODES / 16, H1DIM / 64), 32, 0, stream>>>(
      (const __bf16*)xh, (const __bf16*)Wt, hd, NNODES, FIN, H1DIM);
  k_alphas<<<nwb, 256, 0, stream>>>(hd, as1, ad1, asn, adn, NNODES, H1DIM);

  hipMemsetAsync(smax, 0, (size_t)NNODES * 4, stream);
  hipMemsetAsync(ssum, 0, (size_t)NNODES * 4, stream);
  hipMemsetAsync(agg,  0, (size_t)NNODES * H1DIM * 4, stream);

  k_edge_max<<<eb, 256, 0, stream>>>(ei, asn, adn, smax);
  k_edge_exp<<<eb, 256, 0, stream>>>(ei, asn, adn, smax, ssum, exv);
  k_edge_agg<<<ewb, 256, 0, stream>>>(ei, hd, exv, ssum, agg, H1DIM);
  { int n = NNODES * H1DIM;
    k_bias_act<<<(n + 255) / 256, 256, 0, stream>>>(agg, b1, n, H1DIM, 1); }

  // ============ Layer 2 ============
  { int n8 = (NNODES * H1DIM) / 8;
    k_cvt_bf16x8<<<(n8 + 255) / 256, 256, 0, stream>>>(agg, xh, n8); }   // reuse xh
  { int n = H1DIM * COUT;
    k_transpose_bf16<<<(n + 255) / 256, 256, 0, stream>>>(W2, Wt, H1DIM, COUT); }
  k_wmma_gemm<<<dim3(NNODES / 16, COUT / 64), 32, 0, stream>>>(
      (const __bf16*)xh, (const __bf16*)Wt, hd, NNODES, H1DIM, COUT);   // reuse hd
  k_alphas<<<nwb, 256, 0, stream>>>(hd, as2, ad2, asn, adn, NNODES, COUT);

  hipMemsetAsync(smax, 0, (size_t)NNODES * 4, stream);
  hipMemsetAsync(ssum, 0, (size_t)NNODES * 4, stream);
  hipMemsetAsync(out,  0, (size_t)NNODES * COUT * 4, stream);

  k_edge_max<<<eb, 256, 0, stream>>>(ei, asn, adn, smax);
  k_edge_exp<<<eb, 256, 0, stream>>>(ei, asn, adn, smax, ssum, exv);
  k_edge_agg<<<ewb, 256, 0, stream>>>(ei, hd, exv, ssum, out, COUT);
  { int n = NNODES * COUT;
    k_bias_act<<<(n + 255) / 256, 256, 0, stream>>>(out, b2, n, COUT, 0); }
}
